// ConstructAdjMatrix1_25984552141073
// MI455X (gfx1250) — matrix operand order repaired and checked by reading, not compiled, not run on hardware
//
#include <hip/hip_runtime.h>

// ---------------- Types for WMMA ----------------
typedef __attribute__((ext_vector_type(16))) __bf16 v16bf;
typedef __attribute__((ext_vector_type(8)))  float  v8f;

union BFrag { uint4 u4[2]; v16bf v; };

// ---------------- Constants ----------------
#define NC 4000
#define ND 2000
#define FC 2048
#define FD 1024
#define NTOT 6000
#define BN_EPS 1e-5f
#define LDSK 40   // LDS row stride in bf16 elements: 32 + 8 pad = 80B (16B aligned)

// ---------------- fp32 -> bf16 conversion (RNE) ----------------
__device__ __forceinline__ unsigned short f32_to_bf16_rne(float f) {
    unsigned u = __float_as_uint(f);
    unsigned rounding = 0x7FFFu + ((u >> 16) & 1u);
    return (unsigned short)((u + rounding) >> 16);
}

__global__ void cvt_f32_bf16_kernel(const float* __restrict__ in,
                                    unsigned short* __restrict__ out, int n) {
    int i = blockIdx.x * blockDim.x + threadIdx.x;
    if (i < n) out[i] = f32_to_bf16_rne(in[i]);
}

// ---------------- CDNA5 async global->LDS copy (ASYNCcnt-tracked) ----------------
__device__ __forceinline__ void async_copy_b128(unsigned ldsAddr, const void* gptr) {
    asm volatile("global_load_async_to_lds_b128 %0, %1, off"
                 :: "v"(ldsAddr), "v"((unsigned long long)(uintptr_t)gptr)
                 : "memory");
}
__device__ __forceinline__ void wait_async0() {
    asm volatile("s_wait_asynccnt 0x0" ::: "memory");
}

// ---------------- bf16 WMMA GEMM:  Out[M,N] = A[M,K] @ Bw[N,K]^T + bias[N] ----------------
// Block = 256 threads = 8 waves (2x4). Block tile 64x128, wave tile 32x32 (2x2 WMMA tiles).
// Double-buffered LDS staging filled by GLOBAL_LOAD_ASYNC_TO_LDS_B128.
__global__ __launch_bounds__(256) void gemm_bf16_wmma_lds(
    const unsigned short* __restrict__ A,   // bf16 bits, [M,K] row-major
    const unsigned short* __restrict__ Bw,  // bf16 bits, [N,K] row-major
    const float* __restrict__ bias,         // [N]
    float* __restrict__ Out,                // [M,N] row-major
    int M, int N, int K)
{
    __shared__ unsigned short smA[2][64 * LDSK];    // 2 x 5120 B
    __shared__ unsigned short smB[2][128 * LDSK];   // 2 x 10240 B

    const int t    = threadIdx.x;
    const int lane = t & 31;
    const int wave = t >> 5;
    const int bm = blockIdx.y * 64;
    const int bn = blockIdx.x * 128;

    // ---- cooperative async-copy indices (per thread: 1 A chunk + 2 B chunks of 16B) ----
    const int crow = t >> 2;             // 0..63
    const int ckc  = (t & 3) * 8;        // 0,8,16,24 (bf16 elements)
    const int arow_g = (bm + crow < M) ? (bm + crow) : (M - 1);          // clamped (store-guarded later)
    const int brow0  = crow;             // 0..63
    const int brow1  = crow + 64;        // 64..127
    const int brow0_g = (bn + brow0 < N) ? (bn + brow0) : (N - 1);
    const int brow1_g = (bn + brow1 < N) ? (bn + brow1) : (N - 1);
    const unsigned short* aG  = A  + (size_t)arow_g  * K + ckc;
    const unsigned short* bG0 = Bw + (size_t)brow0_g * K + ckc;
    const unsigned short* bG1 = Bw + (size_t)brow1_g * K + ckc;
    const int ldsAOff = crow  * LDSK + ckc;
    const int ldsB0Off = brow0 * LDSK + ckc;
    const int ldsB1Off = brow1 * LDSK + ckc;

    auto issue_copy = [&](int buf, int k) {
        async_copy_b128((unsigned)(uintptr_t)&smA[buf][ldsAOff],  aG  + k);
        async_copy_b128((unsigned)(uintptr_t)&smB[buf][ldsB0Off], bG0 + k);
        async_copy_b128((unsigned)(uintptr_t)&smB[buf][ldsB1Off], bG1 + k);
    };

    // ---- per-wave fragment addressing in LDS ----
    // A frag (16-bit 16x32): lanes 0-15 row=fr, K{0..7,16..23}; lanes 16-31 row=fr, K{8..15,24..31}
    // B frag (32x16): lanes 0-15 col=fr K{0..7,16..23}; lanes 16-31 col=fr K{8..15,24..31}
    const int wm = (wave >> 2) * 32;     // 0 / 32
    const int wn = (wave & 3) * 32;      // 0 / 32 / 64 / 96
    const int fr  = lane & 15;
    const int ak0 = (lane >> 4) * 8;
    const int bk0 = (lane >> 4) * 8;
    const int aoff0 = (wm + fr) * LDSK + ak0;
    const int aoff1 = aoff0 + 16 * LDSK;
    const int boff0 = (wn + fr) * LDSK + bk0;
    const int boff1 = boff0 + 16 * LDSK;

    v8f c00 = {}, c01 = {}, c10 = {}, c11 = {};

    // ---- prologue: stage K-tile 0 ----
    issue_copy(0, 0);
    wait_async0();
    __syncthreads();

    const int nk = K >> 5;
    for (int kt = 0; kt < nk; ++kt) {
        const int cur = kt & 1;
        if (kt + 1 < nk) issue_copy(cur ^ 1, (kt + 1) * 32);   // async into idle buffer

        const unsigned short* sA = smA[cur];
        const unsigned short* sB = smB[cur];
        BFrag a0, a1, b0, b1;
        a0.u4[0] = *(const uint4*)(sA + aoff0);
        a0.u4[1] = *(const uint4*)(sA + aoff0 + 16);
        a1.u4[0] = *(const uint4*)(sA + aoff1);
        a1.u4[1] = *(const uint4*)(sA + aoff1 + 16);
        b0.u4[0] = *(const uint4*)(sB + boff0);
        b0.u4[1] = *(const uint4*)(sB + boff0 + 16);
        b1.u4[0] = *(const uint4*)(sB + boff1);
        b1.u4[1] = *(const uint4*)(sB + boff1 + 16);

        c00 = __builtin_amdgcn_wmma_f32_16x16x32_bf16(false, a0.v, false, b0.v, (short)0, c00, false, false);
        c01 = __builtin_amdgcn_wmma_f32_16x16x32_bf16(false, a0.v, false, b1.v, (short)0, c01, false, false);
        c10 = __builtin_amdgcn_wmma_f32_16x16x32_bf16(false, a1.v, false, b0.v, (short)0, c10, false, false);
        c11 = __builtin_amdgcn_wmma_f32_16x16x32_bf16(false, a1.v, false, b1.v, (short)0, c11, false, false);

        if (kt + 1 < nk) wait_async0();     // tile kt+1 resident in LDS (this wave's share)
        __syncthreads();                    // publish across the 8 waves
    }

    // ---- store 4 guarded 16x16 tiles (C/D layout: VGPR r -> M=r / M=r+8 by half-wave) ----
    const int rsub = (lane >> 4) << 3;
    auto store_tile = [&](const v8f& c, int gm, int gn) {
        if (gm < M && gn < N) {
            const int colg = gn + fr;
            const float bv = bias[colg];
            const int rbase = gm + rsub;
            #pragma unroll
            for (int r = 0; r < 8; ++r)
                Out[(size_t)(rbase + r) * N + colg] = c[r] + bv;
        }
    };
    store_tile(c00, bm + wm,      bn + wn);
    store_tile(c01, bm + wm,      bn + wn + 16);
    store_tile(c10, bm + wm + 16, bn + wn);
    store_tile(c11, bm + wm + 16, bn + wn + 16);
}

// ---------------- Column stats -> BN affine coefficients a[j], c0[j] ----------------
__global__ void colstats_kernel(const float* __restrict__ X,
                                const float* __restrict__ gamma,
                                const float* __restrict__ beta,
                                float* __restrict__ a, float* __restrict__ c0,
                                int R, int C) {
    int j = blockIdx.x * blockDim.x + threadIdx.x;
    if (j >= C) return;
    float s = 0.f, q = 0.f;
    for (int i = 0; i < R; ++i) {
        float v = X[(size_t)i * C + j];
        s += v; q += v * v;
    }
    float invR = 1.f / (float)R;
    float mu = s * invR;
    float var = fmaxf(q * invR - mu * mu, 0.f);
    float av = gamma[j] * rsqrtf(var + BN_EPS);
    a[j]  = av;
    c0[j] = beta[j] - av * mu;       // BN(x) = a[j]*x + c0[j]
}

// ---------------- Single-block 1D sum ----------------
__global__ void sum1d_kernel(const float* __restrict__ in, int n, float* __restrict__ out) {
    __shared__ float red[256];
    float s = 0.f;
    for (int i = threadIdx.x; i < n; i += blockDim.x) s += in[i];
    red[threadIdx.x] = s; __syncthreads();
    for (int off = 128; off > 0; off >>= 1) {
        if ((int)threadIdx.x < off) red[threadIdx.x] += red[threadIdx.x + off];
        __syncthreads();
    }
    if (threadIdx.x == 0) *out = red[0];
}

// ---------------- Row-wise  sum_j a[j]*X[row,j]  + csum  (rowsum of BN output) ----------------
__global__ void rowdot_kernel(const float* __restrict__ X, const float* __restrict__ a,
                              const float* __restrict__ csum, float* __restrict__ out, int C) {
    __shared__ float red[256];
    int row = blockIdx.x;
    const float4* xr = (const float4*)(X + (size_t)row * C);
    const float4* a4 = (const float4*)a;
    int C4 = C >> 2;
    float s = 0.f;
    for (int j = threadIdx.x; j < C4; j += blockDim.x) {
        float4 x = xr[j], aa = a4[j];
        s += x.x * aa.x + x.y * aa.y + x.z * aa.z + x.w * aa.w;
    }
    red[threadIdx.x] = s; __syncthreads();
    for (int off = 128; off > 0; off >>= 1) {
        if ((int)threadIdx.x < off) red[threadIdx.x] += red[threadIdx.x + off];
        __syncthreads();
    }
    if (threadIdx.x == 0) out[row] = red[0] + csum[0];
}

// ---------------- Plain row sums (block per row) ----------------
__global__ void rowsum_kernel(const float* __restrict__ X, float* __restrict__ out, int C) {
    __shared__ float red[256];
    int row = blockIdx.x;
    const float4* xr = (const float4*)(X + (size_t)row * C);
    int C4 = C >> 2;
    float s = 0.f;
    for (int j = threadIdx.x; j < C4; j += blockDim.x) {
        float4 x = xr[j];
        s += x.x + x.y + x.z + x.w;
    }
    red[threadIdx.x] = s; __syncthreads();
    for (int off = 128; off > 0; off >>= 1) {
        if ((int)threadIdx.x < off) red[threadIdx.x] += red[threadIdx.x + off];
        __syncthreads();
    }
    if (threadIdx.x == 0) out[row] = red[0];
}

// ---------------- Column sums (thread per column, coalesced) ----------------
__global__ void colsum_kernel(const float* __restrict__ X, float* __restrict__ out,
                              int R, int C) {
    int j = blockIdx.x * blockDim.x + threadIdx.x;
    if (j >= C) return;
    float s = 0.f;
    for (int i = 0; i < R; ++i) s += X[(size_t)i * C + j];
    out[j] = s;
}

// ---------------- Degree / self-loop vectors ----------------
__global__ void vectors_kernel(const float* __restrict__ rowbn_cell,
                               const float* __restrict__ rowbn_drug,
                               const float* __restrict__ rowsum_adj,
                               const float* __restrict__ colsum_adj,
                               const float* __restrict__ beta_c,
                               const float* __restrict__ beta_d,
                               float* __restrict__ d_x, float* __restrict__ d_y,
                               float* __restrict__ d_x1, float* __restrict__ d_y1,
                               float* __restrict__ selfv) {
    int i = blockIdx.x * blockDim.x + threadIdx.x;
    if (i >= NTOT) return;
    if (i < NC) {
        float ra = rowsum_adj[i];
        d_x[i] = rsqrtf(ra + 1.f);
        float rc = rowbn_cell[i] + ra;                 // cell_cat row sum == A row sum
        d_x1[i] = rsqrtf(rc + 1.f);
        selfv[i] = 1.f / (rc + 1.f) + 1.f;
        float ca = (float)NC * beta_c[i] + ra;         // A col sum (colsum(BNcell)=NC*beta)
        d_y1[i] = rsqrtf(ca + 1.f);
    } else {
        int k = i - NC;
        float cadj = colsum_adj[k];
        d_y[k] = rsqrtf(cadj + 1.f);
        float rd = cadj + rowbn_drug[k];               // drug_cat row sum == A row sum
        d_x1[i] = rsqrtf(rd + 1.f);
        selfv[i] = 1.f / (rd + 1.f) + 1.f;
        float cA = cadj + (float)ND * beta_d[k];       // A col sum (colsum(BNdrug)=ND*beta)
        d_y1[i] = rsqrtf(cA + 1.f);
    }
}

// ---------------- agg_cell_lp and agg_drug_lp (transpose pair), float4 ----------------
__global__ void agg12_kernel(const float* __restrict__ adj,
                             const float* __restrict__ d_x, const float* __restrict__ d_y,
                             float* __restrict__ o1, float* __restrict__ o2) {
    int j4 = blockIdx.x * blockDim.x + threadIdx.x;
    int i  = blockIdx.y;
    if (j4 >= ND / 4) return;
    float4 av = ((const float4*)adj)[(size_t)i * (ND / 4) + j4];
    float  dxi = d_x[i];
    float4 dy  = ((const float4*)d_y)[j4];
    float4 v = make_float4(dxi * av.x * dy.x, dxi * av.y * dy.y,
                           dxi * av.z * dy.z, dxi * av.w * dy.w);
    ((float4*)o1)[(size_t)i * (ND / 4) + j4] = v;
    int j = j4 * 4;
    o2[(size_t)(j + 0) * NC + i] = v.x;
    o2[(size_t)(j + 1) * NC + i] = v.y;
    o2[(size_t)(j + 2) * NC + i] = v.z;
    o2[(size_t)(j + 3) * NC + i] = v.w;
}

// ---------------- agg_adj_lp over the 6000x6000 block matrix, float4 ----------------
__global__ void agg_adj_kernel(const float* __restrict__ lin_cell,
                               const float* __restrict__ a_c, const float* __restrict__ c0_c,
                               const float* __restrict__ lin_drug,
                               const float* __restrict__ a_d, const float* __restrict__ c0_d,
                               const float* __restrict__ adj,
                               const float* __restrict__ d_x1, const float* __restrict__ d_y1,
                               float* __restrict__ out) {
    int c4 = blockIdx.x * blockDim.x + threadIdx.x;
    int r  = blockIdx.y;
    if (c4 >= NTOT / 4) return;
    int c = c4 * 4;
    float4 v;
    if (r < NC) {
        if (c < NC) {
            float4 x  = ((const float4*)lin_cell)[(size_t)r * (NC / 4) + c4];
            float4 aa = ((const float4*)a_c)[c4];
            float4 cc = ((const float4*)c0_c)[c4];
            v = make_float4(aa.x * x.x + cc.x, aa.y * x.y + cc.y,
                            aa.z * x.z + cc.z, aa.w * x.w + cc.w);
        } else {
            v = ((const float4*)adj)[(size_t)r * (ND / 4) + ((c - NC) >> 2)];
        }
    } else {
        int rr = r - NC;
        if (c < NC) {
            v.x = adj[(size_t)(c + 0) * ND + rr];
            v.y = adj[(size_t)(c + 1) * ND + rr];
            v.z = adj[(size_t)(c + 2) * ND + rr];
            v.w = adj[(size_t)(c + 3) * ND + rr];
        } else {
            int k4 = (c - NC) >> 2;
            float4 x  = ((const float4*)lin_drug)[(size_t)rr * (ND / 4) + k4];
            float4 aa = ((const float4*)a_d)[k4];
            float4 cc = ((const float4*)c0_d)[k4];
            v = make_float4(aa.x * x.x + cc.x, aa.y * x.y + cc.y,
                            aa.z * x.z + cc.z, aa.w * x.w + cc.w);
        }
    }
    float  dr = d_x1[r];
    float4 dc = ((const float4*)d_y1)[c4];
    float4 o = make_float4(dr * v.x * dc.x, dr * v.y * dc.y,
                           dr * v.z * dc.z, dr * v.w * dc.w);
    ((float4*)out)[(size_t)r * (NTOT / 4) + c4] = o;
}

// ---------------- Diagonal matrix fill (zero elsewhere), float4 ----------------
__global__ void diag_fill_kernel(float* __restrict__ out, const float* __restrict__ vals, int n) {
    int c4 = blockIdx.x * blockDim.x + threadIdx.x;
    int r  = blockIdx.y;
    int n4 = n >> 2;
    if (c4 >= n4) return;
    float4 z = make_float4(0.f, 0.f, 0.f, 0.f);
    int base = c4 << 2;
    if (r >= base && r < base + 4) {
        float dv = vals[r];
        int s = r - base;
        if (s == 0) z.x = dv; else if (s == 1) z.y = dv;
        else if (s == 2) z.z = dv; else z.w = dv;
    }
    ((float4*)out)[(size_t)r * n4 + c4] = z;
}

// ================= Host side =================
extern "C" void kernel_launch(void* const* d_in, const int* in_sizes, int n_in,
                              void* d_out, int out_size, void* d_ws, size_t ws_size,
                              hipStream_t stream) {
    const float* adj       = (const float*)d_in[0];
    const float* cell_feat = (const float*)d_in[1];
    const float* drug_feat = (const float*)d_in[2];
    const float* W_cell    = (const float*)d_in[3];
    const float* b_cell    = (const float*)d_in[4];
    const float* W_drug    = (const float*)d_in[5];
    const float* b_drug    = (const float*)d_in[6];
    const float* gamma_c   = (const float*)d_in[7];
    const float* beta_c    = (const float*)d_in[8];
    const float* gamma_d   = (const float*)d_in[9];
    const float* beta_d    = (const float*)d_in[10];

    // ---- workspace carve-up (all offsets 256B aligned) ----
    char* ws = (char*)d_ws;
    size_t off = 0;
    auto alloc = [&](size_t bytes) -> void* {
        void* p = ws + off;
        off += (bytes + 255) & ~(size_t)255;
        return p;
    };
    unsigned short* cf_bf = (unsigned short*)alloc((size_t)NC * FC * 2);
    unsigned short* wc_bf = (unsigned short*)alloc((size_t)NC * FC * 2);
    unsigned short* df_bf = (unsigned short*)alloc((size_t)ND * FD * 2);
    unsigned short* wd_bf = (unsigned short*)alloc((size_t)ND * FD * 2);
    float* lin_cell  = (float*)alloc((size_t)NC * NC * 4);
    float* lin_drug  = (float*)alloc((size_t)ND * ND * 4);
    float* a_c       = (float*)alloc(NC * 4);
    float* c0_c      = (float*)alloc(NC * 4);
    float* a_d       = (float*)alloc(ND * 4);
    float* c0_d      = (float*)alloc(ND * 4);
    float* rowsumA   = (float*)alloc(NC * 4);   // adj row sums
    float* colsumA   = (float*)alloc(ND * 4);   // adj col sums
    float* rowbn_c   = (float*)alloc(NC * 4);   // rowsum of BN(cell)
    float* rowbn_d   = (float*)alloc(ND * 4);   // rowsum of BN(drug)
    float* csum_c    = (float*)alloc(256);
    float* csum_d    = (float*)alloc(256);
    float* d_x       = (float*)alloc(NC * 4);
    float* d_y       = (float*)alloc(ND * 4);
    float* d_x1      = (float*)alloc(NTOT * 4);
    float* d_y1      = (float*)alloc(NTOT * 4);
    float* selfv     = (float*)alloc(NTOT * 4);

    // ---- output carve-up (return order) ----
    float* out = (float*)d_out;
    float* o_agg_cell = out;                                  // [4000 x 2000]
    float* o_agg_drug = out + (size_t)NC * ND;                // [2000 x 4000]
    float* o_agg_adj  = o_agg_drug + (size_t)ND * NC;         // [6000 x 6000]
    float* o_self_cell= o_agg_adj  + (size_t)NTOT * NTOT;     // [4000 x 4000]
    float* o_self_drug= o_self_cell + (size_t)NC * NC;        // [2000 x 2000]
    float* o_self_adj = o_self_drug + (size_t)ND * ND;        // [6000 x 6000]

    const int T = 256;
    // 1) fp32 -> bf16 conversions
    cvt_f32_bf16_kernel<<<(NC * FC + T - 1) / T, T, 0, stream>>>(cell_feat, cf_bf, NC * FC);
    cvt_f32_bf16_kernel<<<(NC * FC + T - 1) / T, T, 0, stream>>>(W_cell,    wc_bf, NC * FC);
    cvt_f32_bf16_kernel<<<(ND * FD + T - 1) / T, T, 0, stream>>>(drug_feat, df_bf, ND * FD);
    cvt_f32_bf16_kernel<<<(ND * FD + T - 1) / T, T, 0, stream>>>(W_drug,    wd_bf, ND * FD);

    // 2) LDS double-buffered WMMA GEMMs: lin = feat @ W^T + b
    {
        dim3 g((NC + 127) / 128, (NC + 63) / 64);
        gemm_bf16_wmma_lds<<<g, 256, 0, stream>>>(cf_bf, wc_bf, b_cell, lin_cell, NC, NC, FC);
    }
    {
        dim3 g((ND + 127) / 128, (ND + 63) / 64);
        gemm_bf16_wmma_lds<<<g, 256, 0, stream>>>(df_bf, wd_bf, b_drug, lin_drug, ND, ND, FD);
    }

    // 3) BN column stats -> affine coefficients
    colstats_kernel<<<(NC + T - 1) / T, T, 0, stream>>>(lin_cell, gamma_c, beta_c, a_c, c0_c, NC, NC);
    colstats_kernel<<<(ND + T - 1) / T, T, 0, stream>>>(lin_drug, gamma_d, beta_d, a_d, c0_d, ND, ND);
    sum1d_kernel<<<1, T, 0, stream>>>(c0_c, NC, csum_c);
    sum1d_kernel<<<1, T, 0, stream>>>(c0_d, ND, csum_d);

    // 4) row sums of BN outputs (dot with a[] + sum(c0))
    rowdot_kernel<<<NC, T, 0, stream>>>(lin_cell, a_c, csum_c, rowbn_c, NC);
    rowdot_kernel<<<ND, T, 0, stream>>>(lin_drug, a_d, csum_d, rowbn_d, ND);

    // 5) adj degree sums
    rowsum_kernel<<<NC, T, 0, stream>>>(adj, rowsumA, ND);
    colsum_kernel<<<(ND + T - 1) / T, T, 0, stream>>>(adj, colsumA, NC, ND);

    // 6) all degree / self-loop vectors
    vectors_kernel<<<(NTOT + T - 1) / T, T, 0, stream>>>(rowbn_c, rowbn_d, rowsumA, colsumA,
                                                         beta_c, beta_d,
                                                         d_x, d_y, d_x1, d_y1, selfv);

    // 7) outputs (2-D grids, float4 stores)
    {
        dim3 g((ND / 4 + T - 1) / T, NC);
        agg12_kernel<<<g, T, 0, stream>>>(adj, d_x, d_y, o_agg_cell, o_agg_drug);
    }
    {
        dim3 g((NTOT / 4 + T - 1) / T, NTOT);
        agg_adj_kernel<<<g, T, 0, stream>>>(lin_cell, a_c, c0_c, lin_drug, a_d, c0_d,
                                            adj, d_x1, d_y1, o_agg_adj);
    }
    {
        dim3 g((NC / 4 + T - 1) / T, NC);
        diag_fill_kernel<<<g, T, 0, stream>>>(o_self_cell, selfv, NC);
    }
    {
        dim3 g((ND / 4 + T - 1) / T, ND);
        diag_fill_kernel<<<g, T, 0, stream>>>(o_self_drug, selfv + NC, ND);
    }
    {
        dim3 g((NTOT / 4 + T - 1) / T, NTOT);
        diag_fill_kernel<<<g, T, 0, stream>>>(o_self_adj, selfv, NTOT);
    }
}
